// SimpleTransformer_14293651161457
// MI455X (gfx1250) — matrix operand order repaired and checked by reading, not compiled, not run on hardware
//
#include <hip/hip_runtime.h>
#include <hip/hip_bf16.h>
#include <math.h>

// ---------------- shapes ----------------
#define VOCAB   32000
#define HDIM    1024
#define LAYERS  8
#define BATCH   2
#define SEQ     2048
#define EPS_LN  1e-5f

typedef __attribute__((ext_vector_type(16))) __bf16          v16bf;
typedef __attribute__((ext_vector_type(8)))  float           v8f;
typedef __attribute__((ext_vector_type(8)))  unsigned short  ushort8;
typedef __attribute__((ext_vector_type(16))) unsigned short  ushort16;

// CDNA5 async global->LDS copy path (guarded; falls back to vector loads)
#if __has_builtin(__builtin_amdgcn_global_load_async_to_lds_b128) && \
    __has_builtin(__builtin_amdgcn_s_wait_asynccnt)
#define USE_ASYNC_LDS 1
#else
#define USE_ASYNC_LDS 0
#endif

#if USE_ASYNC_LDS
typedef __attribute__((__vector_size__(4 * sizeof(int)))) int  i128v;
typedef __attribute__((address_space(1))) i128v*               g128p;  // global
typedef __attribute__((address_space(3))) i128v*               l128p;  // LDS
#endif

// ---------------- helpers ----------------
// fp32 -> bf16 with round-to-nearest-even (integer path; the
// cvt_pk_bf16 builtin lowers to truncating v_mov_b16 pairs on gfx1250,
// which is both longer and numerically worse).
__device__ __forceinline__ unsigned short f2bf(float a) {
  unsigned int u = __builtin_bit_cast(unsigned int, a);
  return (unsigned short)((u + 0x7FFFu + ((u >> 16) & 1u)) >> 16);
}
__device__ __forceinline__ unsigned int pack_bf16x2(float a, float b) {
  return (unsigned int)f2bf(a) | ((unsigned int)f2bf(b) << 16);
}
__device__ __forceinline__ v16bf mk_frag2(const unsigned short* p0,
                                          const unsigned short* p1) {
  ushort8 lo = *(const ushort8*)p0;
  ushort8 hi = *(const ushort8*)p1;
  ushort16 v;
#pragma unroll
  for (int i = 0; i < 8; ++i) { v[i] = lo[i]; v[i + 8] = hi[i]; }
  return __builtin_bit_cast(v16bf, v);
}

__device__ __forceinline__ void copy16_g2l(const unsigned short* gsrc,
                                           unsigned short* ldst) {
#if USE_ASYNC_LDS
  __builtin_amdgcn_global_load_async_to_lds_b128(
      (g128p)(void*)const_cast<unsigned short*>(gsrc),
      (l128p)(void*)ldst, 0, 0);
#else
  *(ushort8*)ldst = *(const ushort8*)gsrc;
#endif
}
__device__ __forceinline__ void wait_async_then_barrier() {
#if USE_ASYNC_LDS
  __builtin_amdgcn_s_wait_asynccnt(0);
#endif
  __syncthreads();
}

// ---------------- the one WMMA GEMM ----------------
// C[M,N] = epi(A[M,K]_f32 * Bt^T + bias [+resid])
//   A  : fp32 [M,K] row-major, leading dim lda (converted to bf16 inline)
//   Bt : bf16 [N,K] row-major, leading dim ldb (pre-converted/transposed)
// EPI: 0 = none, 1 = +bias, 2 = relu(+bias), 3 = +bias +resid
// Block tile 128x128, BK=32, double-buffered LDS. 256 threads = 8 waves
// (4 row-groups x 2 col-groups); each wave owns 32x64 -> 8 v_wmma/K-step.
// B tiles stream via global_load_async_to_lds_b128 while WMMAs run on the
// other buffer; one s_wait_asynccnt + barrier per K-step.
// Requires M%128==0, N%128==0, K%32==0 (holds for every call below).
template <int EPI>
__global__ __launch_bounds__(256) void gemm_wmma(
    const float* __restrict__ A, int lda,
    const unsigned short* __restrict__ Bt, int ldb,
    float* C, int ldc, int K,
    const float* __restrict__ bias,
    const float* __restrict__ resid, int ldr)
{
  __shared__ unsigned short ldsA[2][128 * 40];  // [row][k], pad 32->40
  __shared__ unsigned short ldsB[2][128 * 40];  // [n][k],   pad 32->40

  const int tid  = threadIdx.x;
  const int n0   = blockIdx.x * 128;
  const int m0   = blockIdx.y * 128;
  const int lane = tid & 31;
  const int wv   = tid >> 5;
  const int wm   = wv & 3;          // 4 row groups of 32 rows
  const int wn   = wv >> 2;         // 2 col groups of 64 cols
  const int r16  = lane & 15;
  const int half = lane >> 4;

  v8f acc[2][4];
#pragma unroll
  for (int i = 0; i < 2; ++i)
#pragma unroll
    for (int j = 0; j < 4; ++j) acc[i][j] = v8f{};

  // A staging: thread -> 16 consecutive floats of one row
  const int arow = tid >> 1;            // 0..127
  const int akb  = (tid & 1) << 4;      // 0 or 16
  // B staging: 512 x 16B chunks, 2 per thread (even chunk base -> same row)
  const int bn0 = (tid * 2) >> 2;       // 0..127
  const int ko0 = ((tid * 2) & 3) << 3; // 0 or 16

  auto stage = [&](int k0, int buf) {
    // ---- A (128x32 f32 -> bf16, RNE) ----
    const float* srcA = A + (size_t)(m0 + arow) * lda + k0 + akb;
    float4 f0 = *(const float4*)(srcA);
    float4 f1 = *(const float4*)(srcA + 4);
    float4 f2 = *(const float4*)(srcA + 8);
    float4 f3 = *(const float4*)(srcA + 12);
    unsigned int* d = (unsigned int*)&ldsA[buf][arow * 40 + akb];
    d[0] = pack_bf16x2(f0.x, f0.y); d[1] = pack_bf16x2(f0.z, f0.w);
    d[2] = pack_bf16x2(f1.x, f1.y); d[3] = pack_bf16x2(f1.z, f1.w);
    d[4] = pack_bf16x2(f2.x, f2.y); d[5] = pack_bf16x2(f2.z, f2.w);
    d[6] = pack_bf16x2(f3.x, f3.y); d[7] = pack_bf16x2(f3.z, f3.w);
    // ---- B (128x32 bf16, raw async copy on CDNA5) ----
#pragma unroll
    for (int c = 0; c < 2; ++c) {
      const int ko = ko0 + c * 8;       // 0,8 or 16,24 -> same row bn0
      copy16_g2l(Bt + (size_t)(n0 + bn0) * ldb + k0 + ko,
                 &ldsB[buf][bn0 * 40 + ko]);
    }
  };

  const int nk = K >> 5;
  stage(0, 0);
  for (int kt = 0; kt < nk; ++kt) {
    const int cur = kt & 1;
    wait_async_then_barrier();             // tile kt visible; buffer cur^1 free
    if (kt + 1 < nk) stage((kt + 1) << 5, cur ^ 1);

    // ---- fragments (ISA 16-bit A 16x32 / B 32x16 lane layouts) ----
    v16bf af[2], bf[4];
#pragma unroll
    for (int tm = 0; tm < 2; ++tm) {
      const unsigned short* pa =
          &ldsA[cur][(wm * 32 + tm * 16 + r16) * 40 + half * 8];
      af[tm] = mk_frag2(pa, pa + 16);
    }
#pragma unroll
    for (int tn = 0; tn < 4; ++tn) {
      const unsigned short* pb =
          &ldsB[cur][(wn * 64 + tn * 16 + r16) * 40 + half * 8];
      bf[tn] = mk_frag2(pb, pb + 16);
    }
#pragma unroll
    for (int tm = 0; tm < 2; ++tm)
#pragma unroll
      for (int tn = 0; tn < 4; ++tn)
        acc[tm][tn] = __builtin_amdgcn_wmma_f32_16x16x32_bf16(
            false, af[tm], false, bf[tn], (short)0, acc[tm][tn], false, false);
  }

  // ---- branch-free epilogue: VGPR j -> row j+8*half, col lane%16 ----
#pragma unroll
  for (int tm = 0; tm < 2; ++tm) {
#pragma unroll
    for (int tn = 0; tn < 4; ++tn) {
      const int col = n0 + wn * 64 + tn * 16 + r16;
      float bv = 0.0f;
      if constexpr (EPI >= 1) bv = bias[col];
#pragma unroll
      for (int j = 0; j < 8; ++j) {
        const int row = m0 + wm * 32 + tm * 16 + half * 8 + j;
        float v = acc[tm][tn][j];
        if constexpr (EPI >= 1) v += bv;
        if constexpr (EPI == 2) v = fmaxf(v, 0.0f);
        if constexpr (EPI == 3) v += resid[(size_t)row * ldr + col];
        C[(size_t)row * ldc + col] = v;
      }
    }
  }
}

// ---------------- fp32 -> bf16 convert (plain, row layout preserved) ----------------
// dst[n*K + k] = bf16(src[n*src_ld + k]); 8 elements per thread
__global__ __launch_bounds__(256) void cvt_plain(
    const float* __restrict__ src, int src_ld,
    unsigned short* __restrict__ dst, int K)
{
  const size_t idx = ((size_t)blockIdx.x * 256 + threadIdx.x) * 8;
  const size_t n = idx / K;
  const size_t k = idx % K;
  const float* s = src + n * src_ld + k;
  float4 f0 = *(const float4*)(s);
  float4 f1 = *(const float4*)(s + 4);
  unsigned int* d = (unsigned int*)(dst + idx);
  d[0] = pack_bf16x2(f0.x, f0.y); d[1] = pack_bf16x2(f0.z, f0.w);
  d[2] = pack_bf16x2(f1.x, f1.y); d[3] = pack_bf16x2(f1.z, f1.w);
}

// ---------------- fp32 -> bf16 convert + transpose ----------------
// src: fp32 [K rows, N cols], leading dim src_ld  ->  dst: bf16 [N, K]
// 32x32 tiles via LDS; grid = (N/32, K/32), block = (32, 8)
__global__ __launch_bounds__(256) void cvt_transpose(
    const float* __restrict__ src, int src_ld,
    unsigned short* __restrict__ dst, int K)
{
  __shared__ float t[32][33];
  const int n0 = blockIdx.x * 32;
  const int k0 = blockIdx.y * 32;
  const int tx = threadIdx.x, ty = threadIdx.y;
#pragma unroll
  for (int i = 0; i < 4; ++i)
    t[ty * 4 + i][tx] = src[(size_t)(k0 + ty * 4 + i) * src_ld + n0 + tx];
  __syncthreads();
#pragma unroll
  for (int i = 0; i < 4; ++i)
    dst[(size_t)(n0 + ty * 4 + i) * K + k0 + tx] = f2bf(t[tx][ty * 4 + i]);
}

// ---------------- embedding gather ----------------
__global__ __launch_bounds__(256) void embed_gather(
    const int* __restrict__ x, const float* __restrict__ embed,
    float* __restrict__ h)
{
  const int row = blockIdx.x;
  const int tok = x[row];
  const float4* src = (const float4*)(embed + (size_t)tok * HDIM);
  float4* dst = (float4*)(h + (size_t)row * HDIM);
  for (int i = threadIdx.x; i < HDIM / 4; i += blockDim.x) dst[i] = src[i];
}

// ---------------- softmax over rows of length S ----------------
__global__ __launch_bounds__(256) void softmax_rows(
    float* __restrict__ p, int S, float scale)
{
  __shared__ float red[8];
  float* row = p + (size_t)blockIdx.x * S;
  const int tid = threadIdx.x, lane = tid & 31, wv = tid >> 5;

  float m = -3.4e38f;
  for (int i = tid; i < S; i += 256) m = fmaxf(m, row[i] * scale);
#pragma unroll
  for (int off = 16; off; off >>= 1) m = fmaxf(m, __shfl_xor(m, off, 32));
  if (lane == 0) red[wv] = m;
  __syncthreads();
  m = red[0];
#pragma unroll
  for (int w = 1; w < 8; ++w) m = fmaxf(m, red[w]);
  __syncthreads();

  float s = 0.0f;
  for (int i = tid; i < S; i += 256) {
    float e = __expf(row[i] * scale - m);
    row[i] = e;
    s += e;
  }
#pragma unroll
  for (int off = 16; off; off >>= 1) s += __shfl_xor(s, off, 32);
  if (lane == 0) red[wv] = s;
  __syncthreads();
  s = red[0];
#pragma unroll
  for (int w = 1; w < 8; ++w) s += red[w];
  const float inv = 1.0f / s;
  for (int i = tid; i < S; i += 256) row[i] *= inv;
}

// ---------------- layernorm rows of HDIM ----------------
__global__ __launch_bounds__(256) void layernorm_rows(
    const float* __restrict__ h, const float* __restrict__ gamma,
    const float* __restrict__ beta, float* __restrict__ hn)
{
  __shared__ float r1[8], r2[8];
  const float* row = h + (size_t)blockIdx.x * HDIM;
  float* orow = hn + (size_t)blockIdx.x * HDIM;
  const int tid = threadIdx.x, lane = tid & 31, wv = tid >> 5;

  float s = 0.0f, sq = 0.0f;
  for (int i = tid; i < HDIM; i += 256) {
    float v = row[i];
    s += v; sq += v * v;
  }
#pragma unroll
  for (int off = 16; off; off >>= 1) {
    s  += __shfl_xor(s, off, 32);
    sq += __shfl_xor(sq, off, 32);
  }
  if (lane == 0) { r1[wv] = s; r2[wv] = sq; }
  __syncthreads();
  s = 0.0f; sq = 0.0f;
#pragma unroll
  for (int w = 0; w < 8; ++w) { s += r1[w]; sq += r2[w]; }
  const float mu  = s * (1.0f / HDIM);
  const float var = sq * (1.0f / HDIM) - mu * mu;
  const float inv = rsqrtf(var + EPS_LN);
  for (int i = tid; i < HDIM; i += 256)
    orow[i] = (row[i] - mu) * inv * gamma[i] + beta[i];
}

// ---------------- launcher ----------------
extern "C" void kernel_launch(void* const* d_in, const int* in_sizes, int n_in,
                              void* d_out, int out_size, void* d_ws, size_t ws_size,
                              hipStream_t stream) {
  (void)in_sizes; (void)n_in; (void)out_size; (void)ws_size;

  const int* x        = (const int*)d_in[0];
  const float* embed  = (const float*)d_in[1];
  const float* qkv_w  = (const float*)d_in[2];
  const float* qkv_b  = (const float*)d_in[3];
  const float* out_w  = (const float*)d_in[4];
  const float* out_b  = (const float*)d_in[5];
  const float* up_w   = (const float*)d_in[6];
  const float* up_b   = (const float*)d_in[7];
  const float* down_w = (const float*)d_in[8];
  const float* down_b = (const float*)d_in[9];
  const float* gamma  = (const float*)d_in[10];
  const float* beta   = (const float*)d_in[11];
  float* out          = (float*)d_out;

  const int BS = BATCH * SEQ;               // 4096
  const int H3 = 3 * HDIM, H4 = 4 * HDIM;

  // workspace: fp32 activations then bf16 staging
  float* h       = (float*)d_ws;                               // 16.8 MB
  float* qkv     = h + (size_t)BS * HDIM;                      // 50.3 MB
  float* scores  = qkv + (size_t)BS * H3;                      // 33.6 MB
  float* attnout = scores + (size_t)BATCH * SEQ * SEQ;         // 16.8 MB
  unsigned short* wbf = (unsigned short*)(attnout + (size_t)BS * HDIM); // 8.4 MB
  unsigned short* ebf = wbf + (size_t)H4 * HDIM;               // 65.5 MB
  float* mlp = qkv;       // qkv/scores dead during MLP
  float* hn  = attnout;   // dead after attention

  const dim3 blk(256);
  const dim3 blkT(32, 8);

  auto gemm = [&](int EPI, const float* A, int lda, const unsigned short* Bt,
                  int ldb, float* C, int ldc, int M, int N, int K,
                  const float* bias, const float* resid, int ldr) {
    dim3 grid(N / 128, M / 128);
    switch (EPI) {
      case 0: gemm_wmma<0><<<grid, blk, 0, stream>>>(A, lda, Bt, ldb, C, ldc, K, bias, resid, ldr); break;
      case 1: gemm_wmma<1><<<grid, blk, 0, stream>>>(A, lda, Bt, ldb, C, ldc, K, bias, resid, ldr); break;
      case 2: gemm_wmma<2><<<grid, blk, 0, stream>>>(A, lda, Bt, ldb, C, ldc, K, bias, resid, ldr); break;
      default: gemm_wmma<3><<<grid, blk, 0, stream>>>(A, lda, Bt, ldb, C, ldc, K, bias, resid, ldr); break;
    }
  };
  // convert+transpose: src fp32 [K,N] (ld) -> dst bf16 [N,K]
  auto cvtT = [&](const float* src, int ld, unsigned short* dst, int K, int N) {
    cvt_transpose<<<dim3(N / 32, K / 32), blkT, 0, stream>>>(src, ld, dst, K);
  };
  // plain convert: src fp32 [N rows, K cols] (ld) -> dst bf16 [N,K]
  auto cvtP = [&](const float* src, int ld, unsigned short* dst, int K, int N) {
    cvt_plain<<<(unsigned)((size_t)N * K / 8 / 256), blk, 0, stream>>>(src, ld, dst, K);
  };

  // embeddings: gather fp32 h, and bf16 copy for the tied LM head
  embed_gather<<<BS, blk, 0, stream>>>(x, embed, h);
  cvtP(embed, HDIM, ebf, HDIM, VOCAB);

  const float scale = 1.0f / 32.0f;  // 1/sqrt(H)

  for (int l = 0; l < LAYERS; ++l) {
    const float* qw = qkv_w + (size_t)l * HDIM * H3;
    const float* qb = qkv_b + (size_t)l * H3;
    const float* ow = out_w + (size_t)l * HDIM * HDIM;
    const float* ob = out_b + (size_t)l * HDIM;
    const float* uw = up_w + (size_t)l * HDIM * H4;
    const float* ub = up_b + (size_t)l * H4;
    const float* dw = down_w + (size_t)l * H4 * HDIM;
    const float* db = down_b + (size_t)l * HDIM;

    // qkv = h @ qw + qb
    cvtT(qw, H3, wbf, HDIM, H3);
    gemm(1, h, HDIM, wbf, HDIM, qkv, H3, BS, H3, HDIM, qb, nullptr, 0);

    // scores_b = Q_b @ K_b^T
    for (int b = 0; b < BATCH; ++b) {
      const float* q = qkv + (size_t)b * SEQ * H3;
      const float* k = q + HDIM;
      cvtP(k, H3, wbf, HDIM, SEQ);                 // K_b -> bf16 [S,H]
      gemm(0, q, H3, wbf, HDIM, scores + (size_t)b * SEQ * SEQ, SEQ,
           SEQ, SEQ, HDIM, nullptr, nullptr, 0);
    }
    softmax_rows<<<BATCH * SEQ, blk, 0, stream>>>(scores, SEQ, scale);
    // a_b = attn_b @ V_b
    for (int b = 0; b < BATCH; ++b) {
      const float* v = qkv + (size_t)b * SEQ * H3 + 2 * HDIM;
      cvtT(v, H3, wbf, SEQ, HDIM);                 // V_b^T -> bf16 [H,S]
      gemm(0, scores + (size_t)b * SEQ * SEQ, SEQ, wbf, SEQ,
           attnout + (size_t)b * SEQ * HDIM, HDIM, SEQ, HDIM, SEQ,
           nullptr, nullptr, 0);
    }

    // h = h + a @ ow + ob
    cvtT(ow, HDIM, wbf, HDIM, HDIM);
    gemm(3, attnout, HDIM, wbf, HDIM, h, HDIM, BS, HDIM, HDIM, ob, h, HDIM);
    // mlp = relu(h @ uw + ub)
    cvtT(uw, H4, wbf, HDIM, H4);
    gemm(2, h, HDIM, wbf, HDIM, mlp, H4, BS, H4, HDIM, ub, nullptr, 0);
    // h = h + mlp @ dw + db
    cvtT(dw, HDIM, wbf, H4, HDIM);
    gemm(3, mlp, H4, wbf, H4, h, HDIM, BS, HDIM, H4, db, h, HDIM);
  }

  layernorm_rows<<<BS, blk, 0, stream>>>(h, gamma, beta, hn);

  // tied LM head: out = hn @ embed^T
  gemm(0, hn, HDIM, ebf, HDIM, out, VOCAB, BS, VOCAB, HDIM, nullptr, nullptr, 0);
}
